// QuantumSelfAttention_56014963474600
// MI455X (gfx1250) — compile-verified, matrix operand order checked by reading, not run on hardware
//
#include <hip/hip_runtime.h>
#include <hip/hip_bf16.h>

// ---------------------------------------------------------------------------
// QuantumSelfAttention for MI455X (gfx1250, wave32, WMMA bf16 + TDM path)
//
// Shapes (fixed): B=128, S=128, E=1024, n=8 qubits, L=4 layers.
// Heavy GEMMs (X@Wv, attn@V, out@Wo, psi Gram) run on v_wmma_f32_16x16x32_bf16
// with fp32 accumulation. GEMM A-tiles are staged into LDS with the Tensor
// Data Mover (tensor_load_to_lds, TENSORcnt), double-buffered. The 8-qubit
// state-vector sim + softmax are VALU/LDS.
// ---------------------------------------------------------------------------

typedef __bf16 bf16_t;
typedef __attribute__((ext_vector_type(16))) __bf16 v16bf;
typedef __attribute__((ext_vector_type(8)))  __bf16 v8bf;
typedef __attribute__((ext_vector_type(8)))  float  v8f;
typedef __attribute__((ext_vector_type(4)))  unsigned int v4u;
typedef __attribute__((ext_vector_type(8)))  int v8i;
typedef __attribute__((ext_vector_type(4)))  int v4i;

#define QB   8          // qubits
#define QD   256        // 2^QB amplitudes
#define QL   4          // layers
#define BQ   128        // batch
#define SQ   128        // seq
#define EQ   1024       // embed
#define BS   (BQ*SQ)    // 16384 rows

#if __has_builtin(__builtin_amdgcn_tensor_load_to_lds)
#define HAVE_TDM 1
#else
#define HAVE_TDM 0
#endif

static __device__ inline v8f zero8() {
  v8f z = {0.f,0.f,0.f,0.f,0.f,0.f,0.f,0.f};
  return z;
}

// Load a 16x32 bf16 WMMA fragment (A layout; B assumed symmetric from the
// transposed operand). tile points at (row0, k0); ld = row stride in elems.
// Lanes 0-15: row=lane,    k = {0..7, 16..23}
// Lanes 16-31: row=lane-16, k = {8..15, 24..31}
static __device__ __forceinline__ v16bf load_frag(const bf16_t* tile, int ld) {
  const int lane = threadIdx.x & 31;
  const bf16_t* p = tile + (size_t)(lane & 15) * ld + ((lane & 16) ? 8 : 0);
  v8bf lo = *(const v8bf*)(p);        // k-chunk 0
  v8bf hi = *(const v8bf*)(p + 16);   // k-chunk 1
  v16bf r;
#pragma unroll
  for (int i = 0; i < 8; ++i) { r[i] = lo[i]; r[i + 8] = hi[i]; }
  return r;
}

static __device__ __forceinline__ v8f wmma_bf16(v16bf a, v16bf b, v8f c) {
  return __builtin_amdgcn_wmma_f32_16x16x32_bf16(false, a, false, b,
                                                 (short)0, c, false, false);
}

#if HAVE_TDM
// Issue a TDM 2D tile load (bf16 elements): global (tile_k x tile_rows, row
// stride ld_elems) -> LDS (contiguous row-major). D# per CDNA5 ISA ch.8:
//   group0: count=1 | lds_addr | global_addr[56:0] | type=2
//   group1: data_size=1(2B), tensor_dim0/1, tile_dim0/1, tensor_dim0_stride
static __device__ __forceinline__ void tdm_load_2d(const bf16_t* lds_dst,
                                                   const bf16_t* gsrc,
                                                   int tile_k, int tile_rows,
                                                   long tensor_k, long tensor_rows,
                                                   int ld_elems) {
  unsigned lds_addr = (unsigned)(size_t)lds_dst;
  unsigned long long ga = (unsigned long long)(size_t)gsrc;
  v4u g0;
  g0[0] = 1u;                                           // count=1, user mode
  g0[1] = lds_addr;                                     // LDS byte address
  g0[2] = (unsigned)(ga & 0xffffffffu);                 // global_addr lo
  g0[3] = (unsigned)((ga >> 32) & 0x01ffffffu) | 0x80000000u;  // addr hi | type=2
  v8i g1;
  g1[0] = 1 << 16;                                      // data_size=1 (2 bytes)
  g1[1] = (int)((tensor_k & 0xffff) << 16);             // tensor_dim0[15:0]
  g1[2] = (int)(((tensor_k >> 16) & 0xffff) |
                ((tensor_rows & 0xffff) << 16));        // dim0 hi | dim1 lo
  g1[3] = (int)(((tensor_rows >> 16) & 0xffff) |
                ((unsigned)tile_k << 16));              // dim1 hi | tile_dim0
  g1[4] = tile_rows & 0xffff;                           // tile_dim1 (tile_dim2=0)
  g1[5] = ld_elems;                                     // tensor_dim0_stride lo
  g1[6] = 0;                                            // stride0 hi | stride1 lo
  g1[7] = 0;
  v4i z4 = {0, 0, 0, 0};
#if defined(__clang_major__) && (__clang_major__ >= 23)
  v8i z8 = {0, 0, 0, 0, 0, 0, 0, 0};
  __builtin_amdgcn_tensor_load_to_lds(g0, g1, z4, z4, z8, 0);
#else
  __builtin_amdgcn_tensor_load_to_lds(g0, g1, z4, z4, 0);
#endif
}
#endif  // HAVE_TDM

// --------------------------- conversion kernels ----------------------------

__global__ void k_f32_to_bf16(const float* __restrict__ in,
                              bf16_t* __restrict__ out, int n) {
  for (int i = blockIdx.x * blockDim.x + threadIdx.x; i < n;
       i += gridDim.x * blockDim.x)
    out[i] = (bf16_t)in[i];
}

// out[c][r] = in[r][c]  (R x C  ->  C x R), bf16
__global__ void k_transpose_bf16(const float* __restrict__ in,
                                 bf16_t* __restrict__ out, int R, int C) {
  int idx = blockIdx.x * blockDim.x + threadIdx.x;
  if (idx < R * C) {
    int r = idx / C, c = idx % C;
    out[(size_t)c * R + r] = (bf16_t)in[idx];
  }
}

// ------------------------------- qin = X@Wi --------------------------------

__global__ void k_qin(const float* __restrict__ X, const float* __restrict__ Wi,
                      const float* __restrict__ bi, float* __restrict__ qin) {
  __shared__ float red[256 * QB];
  const size_t row = blockIdx.x;
  const float* xr = X + row * EQ;
  float acc[QB];
#pragma unroll
  for (int j = 0; j < QB; ++j) acc[j] = 0.f;
  for (int e = threadIdx.x; e < EQ; e += 256) {
    float xv = xr[e];
    const float* w = Wi + (size_t)e * QB;
#pragma unroll
    for (int j = 0; j < QB; ++j) acc[j] += xv * w[j];
  }
#pragma unroll
  for (int j = 0; j < QB; ++j) red[threadIdx.x * QB + j] = acc[j];
  __syncthreads();
  for (int off = 128; off > 0; off >>= 1) {
    if ((int)threadIdx.x < off)
#pragma unroll
      for (int j = 0; j < QB; ++j)
        red[threadIdx.x * QB + j] += red[(threadIdx.x + off) * QB + j];
    __syncthreads();
  }
  if (threadIdx.x < QB) qin[row * QB + threadIdx.x] = red[threadIdx.x] + bi[threadIdx.x];
}

// ----------------------- 8-qubit state-vector simulator --------------------
// 2 rows per 256-thread block; state (256 complex fp32) in LDS.
// Gate bit position for qubit q is p = 7-q (matches reference reshape).

__global__ void k_qsim(const float* __restrict__ qin,
                       const float* __restrict__ theta,
                       bf16_t* __restrict__ psi_re,
                       bf16_t* __restrict__ psi_im) {
  __shared__ float sre[2][QD];
  __shared__ float sim_[2][QD];
  const int rsel = threadIdx.x >> 7;       // row within block
  const int t2   = threadIdx.x & 127;      // thread within row (128 pairs)
  const size_t row = (size_t)blockIdx.x * 2 + rsel;
  float* RE = sre[rsel];
  float* IM = sim_[rsel];

  RE[t2] = (t2 == 0) ? 1.0f : 0.0f;
  RE[t2 + 128] = 0.0f;
  IM[t2] = 0.0f;
  IM[t2 + 128] = 0.0f;
  __syncthreads();

  auto ry = [&](int p, float ang) {
    float c = __cosf(0.5f * ang), s = __sinf(0.5f * ang);
    int stride = 1 << p;
    int i0 = ((t2 >> p) << (p + 1)) | (t2 & (stride - 1));
    int i1 = i0 | stride;
    float r0 = RE[i0], r1 = RE[i1], m0 = IM[i0], m1 = IM[i1];
    RE[i0] = c * r0 - s * r1; IM[i0] = c * m0 - s * m1;
    RE[i1] = s * r0 + c * r1; IM[i1] = s * m0 + c * m1;
  };
  auto rz = [&](int p, float ang) {
    float c = __cosf(0.5f * ang), s = __sinf(0.5f * ang);
    int stride = 1 << p;
    int i0 = ((t2 >> p) << (p + 1)) | (t2 & (stride - 1));
    int i1 = i0 | stride;
    float r0 = RE[i0], r1 = RE[i1], m0 = IM[i0], m1 = IM[i1];
    RE[i0] = r0 * c + m0 * s; IM[i0] = m0 * c - r0 * s;   // a0 *= conj(e)
    RE[i1] = r1 * c - m1 * s; IM[i1] = m1 * c + r1 * s;   // a1 *= e
  };
  auto cnot = [&](int q) {
    int pc = 7 - q;
    int pt = 7 - ((q + 1) & 7);
    int stride = 1 << pt;
    int i0 = ((t2 >> pt) << (pt + 1)) | (t2 & (stride - 1));
    int i1 = i0 | stride;
    if ((i0 >> pc) & 1) {
      float a = RE[i0]; RE[i0] = RE[i1]; RE[i1] = a;
      float b = IM[i0]; IM[i0] = IM[i1]; IM[i1] = b;
    }
  };

  for (int q = 0; q < QB; ++q) { ry(7 - q, qin[row * QB + q]); __syncthreads(); }
  for (int l = 0; l < QL; ++l) {
    for (int q = 0; q < QB; ++q) {
      ry(7 - q, theta[((l * QB + q) * 2) + 0]); __syncthreads();
      rz(7 - q, theta[((l * QB + q) * 2) + 1]); __syncthreads();
    }
    for (int q = 0; q < QB; ++q) { cnot(q); __syncthreads(); }
  }

  psi_re[row * QD + t2]       = (bf16_t)RE[t2];
  psi_re[row * QD + t2 + 128] = (bf16_t)RE[t2 + 128];
  psi_im[row * QD + t2]       = (bf16_t)IM[t2];
  psi_im[row * QD + t2 + 128] = (bf16_t)IM[t2 + 128];
}

// -------------------- Gram matrix + kernel matrix (WMMA) -------------------
// One wave per 16x16 tile of K[b][s][t].  G = psi psi^H (complex), psi is
// 128x256 per batch.  G_re = rr+ii, G_im = ir-ri (4 bf16 WMMA accumulators).

__global__ void k_gram(const bf16_t* __restrict__ pre,
                       const bf16_t* __restrict__ pim,
                       const float* __restrict__ phi,
                       float* __restrict__ Kout) {
  const int wid  = blockIdx.x * (blockDim.x >> 5) + (threadIdx.x >> 5);
  const int lane = threadIdx.x & 31;
  const int b  = wid >> 6;
  const int ti = (wid >> 3) & 7;
  const int tj = wid & 7;
  const int s0 = ti * 16, t0 = tj * 16;

  const bf16_t* are = pre + ((size_t)b * SQ + s0) * QD;
  const bf16_t* aim = pim + ((size_t)b * SQ + s0) * QD;
  const bf16_t* bre = pre + ((size_t)b * SQ + t0) * QD;
  const bf16_t* bim = pim + ((size_t)b * SQ + t0) * QD;

  v8f rr = zero8(), ii = zero8(), ir = zero8(), ri = zero8();
#pragma unroll
  for (int k0 = 0; k0 < QD; k0 += 32) {
    v16bf ar = load_frag(are + k0, QD);
    v16bf ai = load_frag(aim + k0, QD);
    v16bf br = load_frag(bre + k0, QD);
    v16bf bi = load_frag(bim + k0, QD);
    rr = wmma_bf16(ar, br, rr);
    ii = wmma_bf16(ai, bi, ii);
    ir = wmma_bf16(ai, br, ir);
    ri = wmma_bf16(ar, bi, ri);
  }

  const float invs = 0.35355339059327373f;  // 1/sqrt(8)
  const int n = lane & 15;
  const int mofs = (lane & 16) ? 8 : 0;
  const float phit = phi[t0 + n];
#pragma unroll
  for (int r = 0; r < 8; ++r) {
    int s = s0 + r + mofs;
    float gre = rr[r] + ii[r];
    float gim = ir[r] - ri[r];
    float v = (gre * gre + gim * gim) * invs + __cosf(phit - phi[s]);
    Kout[((size_t)b * SQ + s) * SQ + t0 + n] = v;
  }
}

// ------------------------------ softmax (row=128) --------------------------

__global__ void k_softmax(const float* __restrict__ Kin,
                          float* __restrict__ attnF,
                          bf16_t* __restrict__ attnB) {
  const int row  = blockIdx.x * (blockDim.x >> 5) + (threadIdx.x >> 5);
  const int lane = threadIdx.x & 31;
  const float* p = Kin + (size_t)row * SQ;
  float v[4];
  float mx = -3.4e38f;
#pragma unroll
  for (int j = 0; j < 4; ++j) { v[j] = p[lane + 32 * j]; mx = fmaxf(mx, v[j]); }
#pragma unroll
  for (int m = 16; m > 0; m >>= 1) mx = fmaxf(mx, __shfl_xor(mx, m, 32));
  float sum = 0.f;
#pragma unroll
  for (int j = 0; j < 4; ++j) { v[j] = __expf(v[j] - mx); sum += v[j]; }
#pragma unroll
  for (int m = 16; m > 0; m >>= 1) sum += __shfl_xor(sum, m, 32);
  float inv = 1.0f / sum;
#pragma unroll
  for (int j = 0; j < 4; ++j) {
    float a = v[j] * inv;
    attnF[(size_t)row * SQ + lane + 32 * j] = a;
    attnB[(size_t)row * SQ + lane + 32 * j] = (bf16_t)a;
  }
}

// ------------------------------ bf16 WMMA GEMM -----------------------------
// C[M][N] = A[M][K] * BT[N][K]^T (+bias).  Block = 256 threads = 8 waves,
// wave grid 2x4, each wave a 16(M) x 64(N) tile (4 accumulators).
// A block tile (32 x 128k bf16, 8KB) is staged into LDS by the Tensor Data
// Mover, double-buffered (wave 0 issues, s_wait_tensorcnt + barrier gates
// the consumers).  B fragments come from global (L2-resident).
// MODE 0: fp32 row-major out  (y)
// MODE 1: bf16 out stored transposed per 128-row batch: Vt[b][n][s]  (V)
// MODE 2: bf16 row-major out  (out)
// bt_stride != 0: B advances per 128-row batch (attn @ V).

#define CK 128   // k-chunk staged per TDM transfer

template <int MODE>
__global__ void k_gemm_bf16(const bf16_t* __restrict__ A,
                            const bf16_t* __restrict__ BT,
                            const float* __restrict__ bias,
                            float* __restrict__ Cf, bf16_t* __restrict__ Cb,
                            int Kd, int lda, int ldbt, long bt_stride,
                            int n_total, int m_total) {
  __shared__ bf16_t aTile[2][32 * CK];   // 2 x 8KB double buffer
  const int wave = threadIdx.x >> 5;
  const int lane = threadIdx.x & 31;
  const int rowB0 = blockIdx.y * 32;                    // block row base
  const int row0  = rowB0 + (wave >> 2) * 16;           // wave row base
  const int col0  = blockIdx.x * 256 + (wave & 3) * 64; // wave col base

  const bf16_t* bt = BT + (size_t)(row0 >> 7) * (size_t)bt_stride;
  const bf16_t* ablk = A + (size_t)rowB0 * lda;

  v8f acc[4] = {zero8(), zero8(), zero8(), zero8()};
  const int nchunk = Kd / CK;

#if HAVE_TDM
  if (wave == 0)
    tdm_load_2d(&aTile[0][0], ablk, CK, 32, Kd, m_total, lda);
  for (int c = 0; c < nchunk; ++c) {
    if (wave == 0) {
      if (c + 1 < nchunk) {
        tdm_load_2d(&aTile[(c + 1) & 1][0], ablk + (size_t)(c + 1) * CK,
                    CK, 32, Kd, m_total, lda);
        __builtin_amdgcn_s_wait_tensorcnt(1);
      } else {
        __builtin_amdgcn_s_wait_tensorcnt(0);
      }
    }
    __syncthreads();
    const bf16_t* aL = &aTile[c & 1][0] + (wave >> 2) * 16 * CK;
    const int kbase = c * CK;
#pragma unroll
    for (int kk = 0; kk < CK; kk += 32) {
      v16bf af = load_frag(aL + kk, CK);     // ds_load_b128 from LDS
#pragma unroll
      for (int i = 0; i < 4; ++i) {
        v16bf bf = load_frag(bt + (size_t)(col0 + 16 * i) * ldbt + kbase + kk,
                             ldbt);
        acc[i] = wmma_bf16(af, bf, acc[i]);
      }
    }
    __syncthreads();
  }
#else
  // Fallback: direct global A fragments (no TDM builtin on this toolchain)
  const bf16_t* a = A + (size_t)row0 * lda;
  for (int k0 = 0; k0 < Kd; k0 += 32) {
    v16bf af = load_frag(a + k0, lda);
#pragma unroll
    for (int i = 0; i < 4; ++i) {
      v16bf bf = load_frag(bt + (size_t)(col0 + 16 * i) * ldbt + k0, ldbt);
      acc[i] = wmma_bf16(af, bf, acc[i]);
    }
  }
#endif

  const int n_lo = lane & 15;
  const int mofs = (lane & 16) ? 8 : 0;
#pragma unroll
  for (int i = 0; i < 4; ++i) {
    int col = col0 + 16 * i + n_lo;
    float bsv = bias ? bias[col] : 0.f;
#pragma unroll
    for (int r = 0; r < 8; ++r) {
      int rowg = row0 + r + mofs;
      float v = acc[i][r] + bsv;
      if (MODE == 0) {
        Cf[(size_t)rowg * n_total + col] = v;
      } else if (MODE == 1) {
        int bb = rowg >> 7, ss = rowg & 127;
        Cb[(size_t)bb * n_total * SQ + (size_t)col * SQ + ss] = (bf16_t)v;
      } else {
        Cb[(size_t)rowg * n_total + col] = (bf16_t)v;
      }
    }
  }
}

// ------------------------------- launcher ----------------------------------

extern "C" void kernel_launch(void* const* d_in, const int* in_sizes, int n_in,
                              void* d_out, int out_size, void* d_ws, size_t ws_size,
                              hipStream_t stream) {
  const float* X     = (const float*)d_in[0];
  const float* Wi    = (const float*)d_in[1];
  const float* bi    = (const float*)d_in[2];
  const float* Wv    = (const float*)d_in[3];
  const float* bv    = (const float*)d_in[4];
  const float* Wo    = (const float*)d_in[5];
  const float* bo    = (const float*)d_in[6];
  const float* theta = (const float*)d_in[7];
  const float* phi   = (const float*)d_in[8];

  float* y    = (float*)d_out;                    // 16384*1024 fp32
  float* attn = (float*)d_out + (size_t)BS * EQ;  // 128*128*128 fp32

  // workspace layout (bytes)
  char* w = (char*)d_ws;
  const size_t OFF_QIN   = 0;                       // 16384*8  f32   = 512KB
  const size_t OFF_PRE   = OFF_QIN  + 524288;       // 16384*256 bf16 = 8MB
  const size_t OFF_PIM   = OFF_PRE  + 8388608;      // 8MB
  const size_t OFF_KMAT  = OFF_PIM  + 8388608;      // 128^3 f32      = 8MB
  const size_t OFF_ATTNB = OFF_KMAT + 8388608;      // 16384*128 bf16 = 4MB
  const size_t OFF_XBF   = OFF_ATTNB + 4194304;     // 16384*1024 bf16 = 32MB (aliased by out_bf)
  const size_t OFF_WVT   = OFF_XBF  + 33554432;     // 1024*1024 bf16 = 2MB
  const size_t OFF_WOT   = OFF_WVT  + 2097152;      // 2MB
  const size_t OFF_VT    = OFF_WOT  + 2097152;      // 128*1024*128 bf16 = 32MB

  float*  qin     = (float*)(w + OFF_QIN);
  bf16_t* psi_re  = (bf16_t*)(w + OFF_PRE);
  bf16_t* psi_im  = (bf16_t*)(w + OFF_PIM);
  float*  Kmat    = (float*)(w + OFF_KMAT);
  bf16_t* attn_bf = (bf16_t*)(w + OFF_ATTNB);
  bf16_t* X_bf    = (bf16_t*)(w + OFF_XBF);
  bf16_t* out_bf  = (bf16_t*)(w + OFF_XBF);   // alias: X_bf dead after V GEMM
  bf16_t* WvT     = (bf16_t*)(w + OFF_WVT);
  bf16_t* WoT     = (bf16_t*)(w + OFF_WOT);
  bf16_t* Vt      = (bf16_t*)(w + OFF_VT);

  // 1) precision conversions
  k_f32_to_bf16<<<4096, 256, 0, stream>>>(X, X_bf, BS * EQ);
  k_transpose_bf16<<<4096, 256, 0, stream>>>(Wv, WvT, EQ, EQ);
  k_transpose_bf16<<<4096, 256, 0, stream>>>(Wo, WoT, EQ, EQ);

  // 2) qin = X @ Wi + bi
  k_qin<<<BS, 256, 0, stream>>>(X, Wi, bi, qin);

  // 3) quantum state simulation -> psi (bf16 re/im planes)
  k_qsim<<<BS / 2, 256, 0, stream>>>(qin, theta, psi_re, psi_im);

  // 4) K = |psi psi^H|^2 / sqrt(8) + cos(phi_t - phi_s)   (WMMA Gram)
  k_gram<<<1024, 256, 0, stream>>>(psi_re, psi_im, phi, Kmat);

  // 5) softmax rows -> attn (fp32 to d_out, bf16 copy for GEMM)
  k_softmax<<<BS / 8, 256, 0, stream>>>(Kmat, attn, attn_bf);

  // 6) V = X @ Wv + bv  (store transposed per batch for B-fragments)
  {
    dim3 grid(EQ / 256, BS / 32);
    k_gemm_bf16<1><<<grid, 256, 0, stream>>>(X_bf, WvT, bv, nullptr, Vt,
                                             EQ, EQ, EQ, 0L, EQ, BS);
  }
  // 7) out = attn @ V   (B advances per 128-row batch; K=128)
  {
    dim3 grid(EQ / 256, BS / 32);
    k_gemm_bf16<2><<<grid, 256, 0, stream>>>(attn_bf, Vt, nullptr, nullptr, out_bf,
                                             SQ, SQ, SQ, (long)EQ * SQ, EQ, BS);
  }
  // 8) y = out @ Wo + bo  (fp32 to d_out)
  {
    dim3 grid(EQ / 256, BS / 32);
    k_gemm_bf16<0><<<grid, 256, 0, stream>>>(out_bf, WoT, bo, y, nullptr,
                                             EQ, EQ, EQ, 0L, EQ, BS);
  }
}
